// RNN_48095043780683
// MI455X (gfx1250) — matrix operand order repaired and checked by reading, not compiled, not run on hardware
//
#include <hip/hip_runtime.h>

// Problem constants
#define B_SZ 4096
#define T_SZ 64
#define I_SZ 60
#define H_SZ 60
#define C_SZ 600
#define CPAD 640      // fc rows padded to 40 WMMA tiles (zero-filled tail)
#define G3   180      // 3*H
#define GP   192      // gate dim padded to 12 WMMA tiles
#define KP   64       // K (=I=H=60) padded to 2 WMMA k-steps
#define K2   7680     // T * 2H  (FC reduction dim)
#define WLD  72       // bf16 LDS row stride (144B, 16B aligned, bank-skewed)
#define GXLD 200      // f32 LDS row stride for gate buffers

typedef __bf16 bf16;
typedef __attribute__((ext_vector_type(16))) bf16  v16bf;
typedef __attribute__((ext_vector_type(8)))  bf16  v8bf;
typedef __attribute__((ext_vector_type(8)))  float v8f;

// Compose a 16-element bf16 fragment from two 16-byte chunks.
__device__ inline v16bf ld_frag(const bf16* p0, const bf16* p1) {
  v8bf lo = *(const v8bf*)p0;
  v8bf hi = *(const v8bf*)p1;
  v16bf f;
#pragma unroll
  for (int i = 0; i < 8; ++i) { f[i] = lo[i]; f[i + 8] = hi[i]; }
  return f;
}

// A fragment (16x32 bf16) from row-major [M][K] LDS tile, k-chunk offset ks.
// Lane L: row = L%16; halves 0..7 = K kb..kb+7, halves 8..15 = K 16+kb..23+kb
// where kb = 8 for lanes 16..31 (per CDNA5 ISA 7.12.2 16-bit A layout).
__device__ inline v16bf frag_a(const bf16* base, int ld, int ks, int lane) {
  const bf16* row = base + (lane & 15) * ld + ks;
  int kb = (lane & 16) ? 8 : 0;
  return ld_frag(row + kb, row + 16 + kb);
}

// B fragment (32x16 bf16) built from W stored [N][K] row-major (native layout
// of w_ih/w_hh/fc_w). Lane L: col n = L%16; halves 0..7 = K kb..kb+7, halves
// 8..15 = K 16+kb..23+kb with kb = 8 for lanes 16..31.
__device__ inline v16bf frag_b(const bf16* base, int ld, int ks, int lane) {
  const bf16* row = base + (lane & 15) * ld + ks;
  int kb = (lane & 16) ? 8 : 0;
  return ld_frag(row + kb, row + 16 + kb);
}

__device__ inline float sigf(float v) { return 1.f / (1.f + __expf(-v)); }
__device__ inline float tanh_f(float v) {
  float t = __expf(-2.f * fabsf(v));
  float r = (1.f - t) / (1.f + t);
  return v < 0.f ? -r : r;
}

// CDNA5 async global->LDS DMA (tracked by ASYNCcnt). lds_off = byte offset of
// LDS destination, goff = 32-bit byte offset added to the SGPR base address.
__device__ inline void async_ld16(unsigned lds_off, unsigned goff, const void* base) {
  asm volatile("global_load_async_to_lds_b128 %0, %1, %2"
               :: "v"(lds_off), "v"(goff), "s"(base)
               : "memory");
}
__device__ inline void wait_async0() {
  asm volatile("s_wait_asynccnt 0x0" ::: "memory");
}

// ---------------------------------------------------------------------------
// Kernel 0: one-shot fc_w f32 -> bf16 conversion into a [640][7680] buffer
// (rows 600..639 zeroed) so the FC kernel can stage B tiles with async DMA
// and needs no bounds checks.
// ---------------------------------------------------------------------------
__global__ void __launch_bounds__(256)
cvt_kernel(const float* __restrict__ fc_w, bf16* __restrict__ fc_wb)
{
  size_t e0 = ((size_t)blockIdx.x * 256 + threadIdx.x) * 8;  // 8 elems/thread
  if (e0 >= (size_t)CPAD * K2) return;
  size_t row = e0 / K2;  // K2 % 8 == 0, so a chunk never crosses rows
  v8bf o;
  if (row < C_SZ) {
#pragma unroll
    for (int i = 0; i < 8; ++i) o[i] = (bf16)fc_w[e0 + i];
  } else {
#pragma unroll
    for (int i = 0; i < 8; ++i) o[i] = (bf16)0.f;
  }
  *(v8bf*)(fc_wb + e0) = o;
}

// ---------------------------------------------------------------------------
// Kernel 1: fused bidirectional GRU. grid = (B/16, 2 directions), block = 128.
// Each block owns a 16-row batch tile; weights live in LDS; per timestep the
// input GEMM (x_t @ Wih^T) and recurrent GEMM (h @ Whh^T) are both done with
// bf16 WMMA (12 n-tiles x 2 k-steps each, 3 n-tiles per wave), then the gate
// nonlinearities run in f32 and the new h is written to LDS (bf16 + f32) and
// to the global hidden buffer consumed by the FC kernel.
// ---------------------------------------------------------------------------
__global__ void __launch_bounds__(128)
gru_kernel(const float* __restrict__ x, const float* __restrict__ w_ih,
           const float* __restrict__ w_hh, const float* __restrict__ b_ih,
           const float* __restrict__ b_hh, bf16* __restrict__ hbuf)
{
  __shared__ __align__(16) bf16 wih_s[GP * WLD];
  __shared__ __align__(16) bf16 whh_s[GP * WLD];
  __shared__ __align__(16) bf16 x_s[16 * WLD];
  __shared__ __align__(16) bf16 h_s[16 * WLD];
  __shared__ float h32_s[16 * KP];
  __shared__ float gx_s[16 * GXLD];
  __shared__ float gh_s[16 * GXLD];
  __shared__ float bih_s[GP];
  __shared__ float bhh_s[GP];

  const int tid  = threadIdx.x;
  const int lane = tid & 31;
  const int wid  = tid >> 5;
  const int d    = blockIdx.y;        // direction: 0 = fwd, 1 = bwd
  const int b0   = blockIdx.x * 16;   // batch tile origin

  // Zero-fill (covers K/N padding and h0 = 0)
  for (int i = tid; i < GP * WLD; i += 128) { wih_s[i] = (bf16)0.f; whh_s[i] = (bf16)0.f; }
  for (int i = tid; i < 16 * WLD; i += 128) { x_s[i] = (bf16)0.f; h_s[i] = (bf16)0.f; }
  for (int i = tid; i < 16 * KP;  i += 128) h32_s[i] = 0.f;
  __syncthreads();

  const float* wih_g = w_ih + (size_t)d * G3 * I_SZ;
  const float* whh_g = w_hh + (size_t)d * G3 * H_SZ;
  for (int i = tid; i < G3 * I_SZ; i += 128) {
    int g = i / I_SZ, k = i % I_SZ;
    wih_s[g * WLD + k] = (bf16)wih_g[i];
    whh_s[g * WLD + k] = (bf16)whh_g[i];
  }
  for (int i = tid; i < GP; i += 128) {
    bih_s[i] = (i < G3) ? b_ih[d * G3 + i] : 0.f;
    bhh_s[i] = (i < G3) ? b_hh[d * G3 + i] : 0.f;
  }
  __syncthreads();

  for (int s = 0; s < T_SZ; ++s) {
    const int tt = d ? (T_SZ - 1 - s) : s;

    // Stage x_t tile (16 x 60 f32 -> bf16)
    for (int i = tid; i < 16 * I_SZ; i += 128) {
      int m = i / I_SZ, k = i % I_SZ;
      x_s[m * WLD + k] =
          (bf16)x[(size_t)(b0 + m) * (T_SZ * I_SZ) + (size_t)tt * I_SZ + k];
    }
    __syncthreads();

    // WMMA phase: each wave owns 3 of the 12 gate-column tiles.
#pragma unroll
    for (int j = 0; j < 3; ++j) {
      const int n0 = (wid * 3 + j) * 16;
      const bf16* bi = wih_s + n0 * WLD;
      const bf16* bh = whh_s + n0 * WLD;
      v8f cx = {};
      v8f ch = {};
      cx = __builtin_amdgcn_wmma_f32_16x16x32_bf16(
          false, frag_a(x_s, WLD, 0, lane), false, frag_b(bi, WLD, 0, lane),
          (short)0, cx, false, false);
      cx = __builtin_amdgcn_wmma_f32_16x16x32_bf16(
          false, frag_a(x_s, WLD, 32, lane), false, frag_b(bi, WLD, 32, lane),
          (short)0, cx, false, false);
      ch = __builtin_amdgcn_wmma_f32_16x16x32_bf16(
          false, frag_a(h_s, WLD, 0, lane), false, frag_b(bh, WLD, 0, lane),
          (short)0, ch, false, false);
      ch = __builtin_amdgcn_wmma_f32_16x16x32_bf16(
          false, frag_a(h_s, WLD, 32, lane), false, frag_b(bh, WLD, 32, lane),
          (short)0, ch, false, false);

      const int nc = n0 + (lane & 15);
      const int mr = (lane >> 4) * 8;
      const float bx = bih_s[nc];
      const float bh2 = bhh_s[nc];
#pragma unroll
      for (int v = 0; v < 8; ++v) {
        gx_s[(mr + v) * GXLD + nc] = cx[v] + bx;   // x@Wih^T + b_ih
        gh_s[(mr + v) * GXLD + nc] = ch[v] + bh2;  // h@Whh^T + b_hh
      }
    }
    __syncthreads();

    // GRU cell elementwise (f32), h carried in f32; bf16 copy feeds next GEMM.
    for (int i = tid; i < 16 * H_SZ; i += 128) {
      int m = i / H_SZ, jx = i % H_SZ;
      float r = sigf(gx_s[m * GXLD + jx] + gh_s[m * GXLD + jx]);
      float z = sigf(gx_s[m * GXLD + 60 + jx] + gh_s[m * GXLD + 60 + jx]);
      float n = tanh_f(gx_s[m * GXLD + 120 + jx] + r * gh_s[m * GXLD + 120 + jx]);
      float hp = h32_s[m * KP + jx];
      float hn = (1.f - z) * n + z * hp;
      h32_s[m * KP + jx] = hn;
      h_s[m * WLD + jx] = (bf16)hn;
      hbuf[(size_t)(b0 + m) * K2 + (size_t)tt * (2 * H_SZ) + d * H_SZ + jx] =
          (bf16)hn;
    }
    __syncthreads();
  }
}

// ---------------------------------------------------------------------------
// Kernel 2: FC head. out[4096,600] = hbuf[4096,7680](bf16) @ fc_wb^T + fc_b.
// grid = (64, 10), block = 256 (8 waves). Block tile 64x64; waves arranged
// 2x4, each wave owns two 16x16 output tiles sharing one B fragment. Both A
// and B tiles are staged into double-buffered LDS with async global->LDS DMA
// (ASYNCcnt) overlapped with the WMMA phase of the previous chunk.
// ---------------------------------------------------------------------------
__global__ void __launch_bounds__(256)
fc_kernel(const bf16* __restrict__ hbuf, const bf16* __restrict__ fc_wb,
          const float* __restrict__ fc_b, float* __restrict__ out)
{
  __shared__ __align__(16) bf16 As[2][64 * WLD];
  __shared__ __align__(16) bf16 Bs[2][64 * WLD];

  const int tid  = threadIdx.x;
  const int lane = tid & 31;
  const int wid  = tid >> 5;
  const int wr   = wid >> 2;  // 0..1 : 32-row slice
  const int wc   = wid & 3;   // 0..3 : 16-col tile
  const int mbase = blockIdx.x * 64;
  const int nbase = blockIdx.y * 64;

  // Staging ownership: 64x64 bf16 tile = 256 threads x 2 b128 transfers.
  const int srow = tid >> 2;            // 0..63
  const int scol = (tid & 3) * 16;      // 0,16,32,48
  const unsigned aoff0 = (unsigned)(((size_t)(mbase + srow) * K2 + scol) * 2);
  const unsigned boff0 = (unsigned)(((size_t)(nbase + srow) * K2 + scol) * 2);
  const unsigned asBase = (unsigned)(size_t)(void*)&As[0][0];
  const unsigned bsBase = (unsigned)(size_t)(void*)&Bs[0][0];
  const unsigned bufBytes = 64 * WLD * 2;
  const unsigned sdst = (unsigned)((srow * WLD + scol) * 2);

  auto stage = [&](int kk, int buf) {
    unsigned kb = (unsigned)(kk * 64 * 2);
    unsigned al = asBase + (unsigned)buf * bufBytes + sdst;
    unsigned bl = bsBase + (unsigned)buf * bufBytes + sdst;
    async_ld16(al,      aoff0 + kb,      hbuf);
    async_ld16(al + 16, aoff0 + kb + 16, hbuf);
    async_ld16(bl,      boff0 + kb,      fc_wb);
    async_ld16(bl + 16, boff0 + kb + 16, fc_wb);
  };

  stage(0, 0);
  wait_async0();
  __syncthreads();

  v8f acc0 = {};
  v8f acc1 = {};
  const int NKK = K2 / 64;  // 120 chunks
  for (int kk = 0; kk < NKK; ++kk) {
    const int cur = kk & 1;
    if (kk + 1 < NKK) stage(kk + 1, cur ^ 1);  // DMA overlaps WMMA below

    const bf16* a0 = &As[cur][(wr * 32) * WLD];
    const bf16* a1 = &As[cur][(wr * 32 + 16) * WLD];
    const bf16* bb = &Bs[cur][(wc * 16) * WLD];
#pragma unroll
    for (int ks = 0; ks < 2; ++ks) {
      v16bf fb = frag_b(bb, WLD, ks * 32, lane);
      acc0 = __builtin_amdgcn_wmma_f32_16x16x32_bf16(
          false, frag_a(a0, WLD, ks * 32, lane), false, fb,
          (short)0, acc0, false, false);
      acc1 = __builtin_amdgcn_wmma_f32_16x16x32_bf16(
          false, frag_a(a1, WLD, ks * 32, lane), false, fb,
          (short)0, acc1, false, false);
    }

    wait_async0();
    __syncthreads();
  }

  const int ncol = nbase + wc * 16 + (lane & 15);
  if (ncol < C_SZ) {
    const float bb = fc_b[ncol];
    const int mr0 = mbase + wr * 32 + (lane >> 4) * 8;
#pragma unroll
    for (int v = 0; v < 8; ++v)
      out[(size_t)(mr0 + v) * C_SZ + ncol] = acc0[v] + bb;
#pragma unroll
    for (int v = 0; v < 8; ++v)
      out[(size_t)(mr0 + 16 + v) * C_SZ + ncol] = acc1[v] + bb;
  }
}

// ---------------------------------------------------------------------------
extern "C" void kernel_launch(void* const* d_in, const int* in_sizes, int n_in,
                              void* d_out, int out_size, void* d_ws, size_t ws_size,
                              hipStream_t stream)
{
  const float* x    = (const float*)d_in[0];
  const float* w_ih = (const float*)d_in[1];
  const float* w_hh = (const float*)d_in[2];
  const float* b_ih = (const float*)d_in[3];
  const float* b_hh = (const float*)d_in[4];
  const float* fc_w = (const float*)d_in[5];
  const float* fc_b = (const float*)d_in[6];
  float* out = (float*)d_out;

  // Workspace layout: hbuf [4096][7680] bf16 (62.9MB) | fc_wb [640][7680] bf16
  bf16* hbuf  = (bf16*)d_ws;
  bf16* fc_wb = (bf16*)((char*)d_ws + (size_t)B_SZ * K2 * 2);

  dim3 gc((CPAD * K2 / 8 + 255) / 256), bc(256);
  cvt_kernel<<<gc, bc, 0, stream>>>(fc_w, fc_wb);

  dim3 g1(B_SZ / 16, 2), b1(128);
  gru_kernel<<<g1, b1, 0, stream>>>(x, w_ih, w_hh, b_ih, b_hh, hbuf);

  dim3 g2(B_SZ / 64, CPAD / 64), b2(256);
  fc_kernel<<<g2, b2, 0, stream>>>(hbuf, fc_wb, fc_b, out);
}
